// SingleLevelTextUpdateV3_85959475462310
// MI455X (gfx1250) — matrix operand order repaired and checked by reading, not compiled, not run on hardware
//
#include <hip/hip_runtime.h>
#include <hip/hip_bf16.h>

// ---------------- problem constants ----------------
#define B_    16
#define C_    256
#define HW_   16384     // 128*128
#define NCLS  80
#define TDIM  512
#define HID   256       // == C_

// ---------------- phase-1 tiling ----------------
#define NCHUNK 16                 // pixel chunks per batch -> 256 workgroups
#define PCHUNK (HW_ / NCHUNK)     // 1024 pixels per workgroup
#define TP     128                // pixels per tile
#define NT     5                  // 80 classes = 5 tiles of 16
// padded LDS strides (in halves); +8 halves => bank-conflict-free frag reads
#define XSTR   (C_ + 8)           // 264  x_lds  [TP][XSTR]
#define XTSTR  (TP + 8)           // 136  xT_lds [C_][XTSTR]
#define QSTR   (C_ + 8)           // 264  qk_lds [NCLS][QSTR]
#define WSTR   (TP + 8)           // 136  wT_lds [NCLS][WSTR]

typedef __attribute__((ext_vector_type(16))) _Float16 v16h;
typedef __attribute__((ext_vector_type(8)))  _Float16 v8h;
typedef __attribute__((ext_vector_type(8)))  float    v8f;

// A-matrix fragment (16x32 f16): lane L(0-15) row m=L: K = 0..7 & 16..23 ;
// lanes 16-31: K = 8..15 & 24..31.  src is row-major [m][k].
__device__ __forceinline__ v16h load_frag_a(const _Float16* row, int k0, int hi8) {
  union { v16h v; v8h h[2]; } u;
  u.h[0] = *(const v8h*)(row + k0 + hi8);
  u.h[1] = *(const v8h*)(row + k0 + 16 + hi8);
  return u.v;
}
// B-matrix fragment (32x16 f16): lane L(0-15) col n=L holds K=0..7 & 16..23,
// lanes 16-31 hold K=8..15 & 24..31.  src is the TRANSPOSED matrix, row-major [n][k].
__device__ __forceinline__ v16h load_frag_b(const _Float16* row, int k0, int hi16) {
  union { v16h v; v8h h[2]; } u;
  const _Float16* p = row + k0 + hi16 / 2;
  u.h[0] = *(const v8h*)(p);
  u.h[1] = *(const v8h*)(p + 16);
  return u.v;
}

// ---------------- kernel A: Qk = scale*(text@Wq^T+bq)@Wk, sbias ----------------
__global__ void __launch_bounds__(256) prep_qk(
    const float* __restrict__ text, const float* __restrict__ Wq,
    const float* __restrict__ bq,   const float* __restrict__ Wk,
    const float* __restrict__ bk,   _Float16* __restrict__ qk_g,
    float* __restrict__ sbias_g)
{
  __shared__ float Q_s[HID];
  __shared__ float red[HID];
  const int n = blockIdx.x, tid = threadIdx.x;
  const float scale = 0.0625f;  // 1/sqrt(256)
  // Q[n,tid]
  {
    float acc = bq[tid];
    const float* tr = text + (size_t)n * TDIM;
    const float* wr = Wq + (size_t)tid * TDIM;
    for (int d = 0; d < TDIM; ++d) acc += tr[d] * wr[d];
    Q_s[tid] = acc;
  }
  __syncthreads();
  // sbias[n] = scale * dot(Q, bk)
  red[tid] = Q_s[tid] * bk[tid];
  __syncthreads();
  for (int s = 128; s > 0; s >>= 1) {
    if (tid < s) red[tid] += red[tid + s];
    __syncthreads();
  }
  if (tid == 0) sbias_g[n] = scale * red[0];
  // Qk[n,tid] = scale * sum_h Q[h]*Wk[h,tid]
  {
    float acc = 0.f;
    for (int h = 0; h < HID; ++h) acc += Q_s[h] * Wk[(size_t)h * C_ + tid];
    qk_g[(size_t)n * C_ + tid] = (_Float16)(scale * acc);
  }
}

// ---------------- kernel B: fused streaming flash attention over x ----------------
extern __shared__ char smem_raw[];

__global__ void __launch_bounds__(256, 1) attn_phase1(
    const float* __restrict__ fused, const _Float16* __restrict__ qk_g,
    const float* __restrict__ sbias_g,
    float* __restrict__ m_part, float* __restrict__ l_part,
    float* __restrict__ xw_part)
{
  const int b = blockIdx.x / NCHUNK, chunk = blockIdx.x % NCHUNK;
  const int tid = threadIdx.x;
  const int w = tid >> 5, lane = tid & 31;
  const int ln = lane & 15, hi = lane >> 4;

  _Float16* x_lds  = (_Float16*)smem_raw;            // [TP][XSTR]
  _Float16* xT_lds = x_lds + TP * XSTR;              // [C_][XTSTR]
  _Float16* qk_lds = xT_lds + C_ * XTSTR;            // [NCLS][QSTR]
  _Float16* wT_lds = qk_lds + NCLS * QSTR;           // [NCLS][WSTR]
  float* sbias_lds = (float*)(wT_lds + NCLS * WSTR); // [NCLS]
  float* m_run  = sbias_lds + NCLS;                  // [NCLS]
  float* l_run  = m_run + NCLS;                      // [NCLS]
  float* sfac   = l_run + NCLS;                      // [NCLS]
  float* colmax = sfac + NCLS;                       // [8][NCLS]
  float* colsum = colmax + 8 * NCLS;                 // [8][NCLS]

  // stage Qk (f16) + sbias, init running stats
  for (int i = tid; i < NCLS * C_; i += 256) {
    int n = i / C_, c = i - n * C_;
    qk_lds[n * QSTR + c] = qk_g[i];
  }
  if (tid < NCLS) {
    sbias_lds[tid] = sbias_g[tid];
    m_run[tid] = -__builtin_inff();
    l_run[tid] = 0.f;
  }

  v8f xw[2][NT] = {};   // persistent accumulators: wave owns c-tiles {2w,2w+1} x 5 n-tiles

  const size_t base = (size_t)b * C_ * HW_ + (size_t)chunk * PCHUNK;

  for (int t = 0; t < PCHUNK / TP; ++t) {
    const int p0 = t * TP;
    __syncthreads();  // protect x_lds / xT_lds reuse

    // ---- stage x tile: f32 global -> f16 LDS in [p][c] and [c][p] layouts ----
    for (int i = tid; i < C_ * (TP / 4); i += 256) {
      const int c = i >> 5;              // TP/4 == 32
      const int pq = (i & 31) << 2;
      const float4 v = *(const float4*)(fused + base + (size_t)c * HW_ + p0 + pq);
      const _Float16 h0 = (_Float16)v.x, h1 = (_Float16)v.y,
                     h2 = (_Float16)v.z, h3 = (_Float16)v.w;
      x_lds[(pq + 0) * XSTR + c] = h0;
      x_lds[(pq + 1) * XSTR + c] = h1;
      x_lds[(pq + 2) * XSTR + c] = h2;
      x_lds[(pq + 3) * XSTR + c] = h3;
      _Float16* xr = &xT_lds[c * XTSTR + pq];
      xr[0] = h0; xr[1] = h1; xr[2] = h2; xr[3] = h3;
      if (t + 1 < PCHUNK / TP && (i & 31) == 0)
        __builtin_prefetch(fused + base + (size_t)c * HW_ + p0 + TP, 0, 1);
    }
    __syncthreads();

    // ---- scores: S[16w..16w+15][0..79] = x_tile @ Qk^T ----
    v8f sacc[NT] = {};
    const _Float16* arow = &x_lds[(16 * w + ln) * XSTR];
    for (int kc = 0; kc < C_; kc += 32) {
      const v16h a = load_frag_a(arow, kc, hi * 8);
#pragma unroll
      for (int nt = 0; nt < NT; ++nt) {
        const v16h bb = load_frag_b(&qk_lds[(nt * 16 + ln) * QSTR], kc, hi * 16);
        sacc[nt] = __builtin_amdgcn_wmma_f32_16x16x32_f16(
            false, a, false, bb, (short)0, sacc[nt], false, false);
      }
    }

    // ---- per-column (over p) max within this wave's 16 rows ----
#pragma unroll
    for (int nt = 0; nt < NT; ++nt) {
      const float sb = sbias_lds[nt * 16 + ln];
      float mx = -__builtin_inff();
#pragma unroll
      for (int r = 0; r < 8; ++r) {
        sacc[nt][r] += sb;
        mx = fmaxf(mx, sacc[nt][r]);
      }
      mx = fmaxf(mx, __shfl_xor(mx, 16, 32));
      if (hi == 0) colmax[w * NCLS + nt * 16 + ln] = mx;
    }
    __syncthreads();

    // ---- update running max, compute rescale factor ----
    if (tid < NCLS) {
      float tm = colmax[tid];
      for (int ww = 1; ww < 8; ++ww) tm = fmaxf(tm, colmax[ww * NCLS + tid]);
      const float mo = m_run[tid];
      const float mn = fmaxf(mo, tm);
      const float sf = __expf(mo - mn);
      m_run[tid] = mn;
      sfac[tid] = sf;
      l_run[tid] *= sf;
    }
    __syncthreads();

    // ---- weights: e = exp(s - m_new); write transposed f16, accumulate col sums ----
#pragma unroll
    for (int nt = 0; nt < NT; ++nt) {
      const int n = nt * 16 + ln;
      const float mn = m_run[n];
      float rs = 0.f;
      _Float16* wrow = &wT_lds[n * WSTR + 16 * w + hi * 8];
#pragma unroll
      for (int r = 0; r < 8; ++r) {
        const float e = __expf(sacc[nt][r] - mn);
        rs += e;
        wrow[r] = (_Float16)e;
      }
      rs += __shfl_xor(rs, 16, 32);
      if (hi == 0) colsum[w * NCLS + n] = rs;
    }
    __syncthreads();

    if (tid < NCLS) {
      float ts = 0.f;
      for (int ww = 0; ww < 8; ++ww) ts += colsum[ww * NCLS + tid];
      l_run[tid] += ts;
    }

    // ---- xw accumulate: xw[n][c] = sfac*xw + w^T @ x  (WMMA) ----
#pragma unroll
    for (int cti = 0; cti < 2; ++cti) {
      const int ct = 2 * w + cti;
      const _Float16* brow = &xT_lds[(ct * 16 + ln) * XTSTR];
#pragma unroll
      for (int mt = 0; mt < NT; ++mt) {
#pragma unroll
        for (int r = 0; r < 8; ++r)
          xw[cti][mt][r] *= sfac[mt * 16 + hi * 8 + r];
        const _Float16* arow2 = &wT_lds[(mt * 16 + ln) * WSTR];
#pragma unroll
        for (int kp = 0; kp < TP; kp += 32) {
          const v16h a = load_frag_a(arow2, kp, hi * 8);
          const v16h bb = load_frag_b(brow, kp, hi * 16);
          xw[cti][mt] = __builtin_amdgcn_wmma_f32_16x16x32_f16(
              false, a, false, bb, (short)0, xw[cti][mt], false, false);
        }
      }
    }
  }

  // ---- write per-chunk partials ----
  const size_t pb = (size_t)b * NCHUNK + chunk;
  if (tid < NCLS) {
    m_part[pb * NCLS + tid] = m_run[tid];
    l_part[pb * NCLS + tid] = l_run[tid];
  }
#pragma unroll
  for (int cti = 0; cti < 2; ++cti)
#pragma unroll
    for (int mt = 0; mt < NT; ++mt)
#pragma unroll
      for (int r = 0; r < 8; ++r) {
        const int n = mt * 16 + hi * 8 + r;
        const int c = (2 * w + cti) * 16 + ln;
        xw_part[(pb * NCLS + n) * C_ + c] = xw[cti][mt][r];
      }
}

// ---------------- kernel C: merge chunks + Wv/Wo projections + confidence ----------------
__global__ void __launch_bounds__(256) attn_phase2(
    const float* __restrict__ m_part, const float* __restrict__ l_part,
    const float* __restrict__ xw_part,
    const float* __restrict__ Wv, const float* __restrict__ bv,
    const float* __restrict__ Wo, const float* __restrict__ bo,
    const float* __restrict__ conf_bias, float* __restrict__ out)
{
  __shared__ float m_s[NCHUNK], e_s[NCHUNK];
  __shared__ float MS, LS;
  __shared__ float xw_s[C_], ao_s[HID];
  const int bid = blockIdx.x;
  const int b = bid / NCLS, n = bid - b * NCLS;
  const int tid = threadIdx.x;

  if (tid < NCHUNK) m_s[tid] = m_part[((size_t)b * NCHUNK + tid) * NCLS + n];
  __syncthreads();
  if (tid == 0) {
    float M = -__builtin_inff();
    for (int i = 0; i < NCHUNK; ++i) M = fmaxf(M, m_s[i]);
    MS = M;
  }
  __syncthreads();
  if (tid < NCHUNK) e_s[tid] = __expf(m_s[tid] - MS);
  __syncthreads();
  if (tid == 0) {
    float L = 0.f;
    for (int i = 0; i < NCHUNK; ++i)
      L += l_part[((size_t)b * NCHUNK + i) * NCLS + n] * e_s[i];
    LS = L;
  }
  __syncthreads();
  {  // combine weighted x sums, normalize
    const int c = tid;
    float acc = 0.f;
    for (int ch = 0; ch < NCHUNK; ++ch)
      acc += xw_part[(((size_t)b * NCHUNK + ch) * NCLS + n) * C_ + c] * e_s[ch];
    xw_s[c] = acc / LS;
  }
  __syncthreads();
  {  // attn_out = Wv @ xw + bv   (Wv is L2-resident)
    const int h = tid;
    float acc = bv[h];
    const float* wr = Wv + (size_t)h * C_;
    for (int c = 0; c < C_; ++c) acc += wr[c] * xw_s[c];
    ao_s[h] = acc;
  }
  __syncthreads();
  const float conf = 1.f / (1.f + __expf(-(MS + conf_bias[0])));
  for (int o = tid; o < TDIM; o += 256) {
    float acc = bo[o];
    const float* wr = Wo + (size_t)o * HID;
    for (int h = 0; h < HID; ++h) acc += wr[h] * ao_s[h];
    out[((size_t)b * NCLS + n) * TDIM + o] = acc * conf;
  }
}

// ---------------- launcher ----------------
extern "C" void kernel_launch(void* const* d_in, const int* in_sizes, int n_in,
                              void* d_out, int out_size, void* d_ws, size_t ws_size,
                              hipStream_t stream) {
  (void)in_sizes; (void)n_in; (void)out_size; (void)ws_size;
  const float* fused = (const float*)d_in[0];
  const float* text  = (const float*)d_in[1];
  const float* Wq = (const float*)d_in[2];
  const float* bq = (const float*)d_in[3];
  const float* Wk = (const float*)d_in[4];
  const float* bk = (const float*)d_in[5];
  const float* Wv = (const float*)d_in[6];
  const float* bv = (const float*)d_in[7];
  const float* Wo = (const float*)d_in[8];
  const float* bo = (const float*)d_in[9];
  const float* cb = (const float*)d_in[10];
  float* out = (float*)d_out;

  char* ws = (char*)d_ws;
  _Float16* qk_g  = (_Float16*)(ws + 0);            // 80*256*2    = 40960
  float*    sb_g  = (float*)(ws + 40960);           // 80*4        = 320
  float*    m_g   = (float*)(ws + 41472);           // 256*80*4    = 81920
  float*    l_g   = (float*)(ws + 41472 + 81920);   // 81920
  float*    xw_g  = (float*)(ws + 41472 + 163840);  // 256*80*256*4 = 20.97 MB

  prep_qk<<<NCLS, 256, 0, stream>>>(text, Wq, bq, Wk, bk, qk_g, sb_g);

  const size_t smem =
      (size_t)(TP * XSTR + C_ * XTSTR + NCLS * QSTR + NCLS * WSTR) * 2 +
      (size_t)(NCLS * 4 + 16 * NCLS) * 4;  // ~207.6 KB of the 320 KB WGP LDS
  hipFuncSetAttribute((const void*)attn_phase1,
                      hipFuncAttributeMaxDynamicSharedMemorySize, (int)smem);
  attn_phase1<<<B_ * NCHUNK, 256, smem, stream>>>(fused, qk_g, sb_g, m_g, l_g, xw_g);

  attn_phase2<<<B_ * NCLS, 256, 0, stream>>>(m_g, l_g, xw_g, Wv, bv, Wo, bo, cb, out);
}
